// SimpleLSTM_64871186039205
// MI455X (gfx1250) — matrix operand order repaired and checked by reading, not compiled, not run on hardware
//
#include <hip/hip_runtime.h>

// ---------------------------------------------------------------------------
// 3-layer LSTM (B=256, T=512, F=32, U=128) + dense(1) head for gfx1250.
// Grid = 16 WGs (one per 16-row batch tile, the WMMA M dim), 256 threads
// (8 wave32). Recurrence is batch-independent -> zero inter-WG sync.
// Weights (W||U -> f16) live in LDS pre-swizzled into WMMA B-fragment order
// (needs CDNA5's 320KB/WGP LDS). Recurrent GEMM: v_wmma_f32_16x16x32_f16.
// Wave w owns hidden units [16w,16w+16): its 4 N-tiles are exactly the
// i/f/g/o gate columns for those units -> gate math is wave-local.
// Per-step input is double-buffered in LDS (t+1 global load hidden behind the
// WMMA chain). sched_group_barrier directives pipeline the ds_load operand
// groups one K-chunk ahead of the WMMA chain.
// d_ws holds the inter-layer h sequence: [16 WG][T][16][128] f16 = 32 MiB.
// ---------------------------------------------------------------------------

typedef __attribute__((ext_vector_type(16))) _Float16 v16h;
typedef __attribute__((ext_vector_type(8)))  _Float16 v8h;
typedef __attribute__((ext_vector_type(8)))  float    v8f;

#define T_LEN 512
#define F_IN  32
#define U_DIM 128
#define NG    512          // 4*U
#define HST   (U_DIM + 8)  // h-region row stride (f16), keeps rows 16B aligned
#define SLOT  (16 * 136)   // fixed input-slot pitch (max layer row stride)

// sched_group_barrier masks
#define SG_MFMA    0x008   // matrix (WMMA) ops
#define SG_DSREAD  0x100   // LDS reads

__device__ __forceinline__ float rcp_f(float x) {
  return __builtin_amdgcn_rcpf(x);              // v_rcp_f32
}
__device__ __forceinline__ float sigmoid_f(float x) {
  return rcp_f(1.0f + __expf(-x));              // v_exp_f32 + v_rcp_f32
}
__device__ __forceinline__ float tanh_f(float x) {
  float e = __expf(-2.0f * x);
  return (1.0f - e) * rcp_f(1.0f + e);
}
__device__ __forceinline__ v16h mk16(v8h lo, v8h hi) {
  v16h r;
#pragma unroll
  for (int e = 0; e < 8; ++e) { r[e] = lo[e]; r[e + 8] = hi[e]; }
  return r;
}

// One LSTM layer over this WG's 16 batch rows.
// LDS: wlds   = WMMA B fragments of [W;U] (16 f16/lane, 32B contiguous)
//      inBase = 2 double-buffered input slots [16][FIN+8]
//      hBase  = recurrent h tile [16][128+8]
template <int FIN, bool FIRST, bool LAST>
__device__ __forceinline__ void lstm_layer(
    const float* __restrict__ xin,          // FIRST only: x [B,T,32] f32
    _Float16*    __restrict__ hseq,         // ws: [16][T][16][128] f16 (in/out)
    const float* __restrict__ Wg,           // [FIN, 512] f32
    const float* __restrict__ Ug,           // [128, 512] f32
    const float* __restrict__ bgv,          // [512] f32
    _Float16* __restrict__ wlds,
    _Float16* __restrict__ inBase,
    _Float16* __restrict__ hBase)
{
  constexpr int KTOT = FIN + U_DIM;         // 160 or 256
  constexpr int KC   = KTOT / 32;           // 5 or 8 K-chunks
  constexpr int KCI  = FIN / 32;            // input K-chunks (1 or 4)
  constexpr int INST = FIN + 8;             // input slot row stride

  const int tid  = threadIdx.x;
  const int lane = tid & 31;
  const int w    = tid >> 5;                // wave id 0..7 -> hidden tile
  const int wg   = blockIdx.x;
  const int colL = lane & 15;
  const int mOff = (lane & 16) ? 8 : 0;     // C/D layout: lanes 16-31 hold M+8
  const int u    = w * 16 + colL;           // this lane's hidden-unit column
  const int lb   = (lane & 16) ? 8 : 0;     // A layout K sub-base

  // ---- stage f32 weights -> f16 WMMA B-fragments in LDS (once per layer) ----
  for (int idx = tid; idx < KC * 32 * 32; idx += 256) {
    const int l  = idx & 31;
    const int nt = (idx >> 5) & 31;
    const int kc = idx >> 10;
    const int n  = nt * 16 + (l & 15);
    const int kb = kc * 32 + ((l & 16) ? 8 : 0);
    _Float16* dst = wlds + idx * 16;
#pragma unroll
    for (int e = 0; e < 16; ++e) {
      const int k = kb + (e & 7) + ((e & 8) ? 16 : 0);
      const float v = (k < FIN) ? Wg[k * NG + n] : Ug[(k - FIN) * NG + n];
      dst[e] = (_Float16)v;
    }
  }
  // h0 = 0
  for (int idx = tid; idx < 16 * HST; idx += 256) hBase[idx] = (_Float16)0.0f;

  // ---- prefill input slot 0 with t = 0 ----
  if constexpr (FIRST) {
    for (int idx = tid; idx < 16 * 32; idx += 256) {
      const int row = idx >> 5, c = idx & 31;
      inBase[row * INST + c] =
          (_Float16)xin[(((size_t)(wg * 16 + row)) * T_LEN + 0) * F_IN + c];
    }
  } else {
    v8h v = *(const v8h*)(hseq + ((size_t)wg * T_LEN + 0) * 16 * U_DIM + tid * 8);
    *(v8h*)(inBase + (tid >> 4) * INST + (tid & 15) * 8) = v;
  }
  __syncthreads();

  const float bi = bgv[u];
  const float bf = bgv[U_DIM + u];
  const float bg = bgv[2 * U_DIM + u];
  const float bo = bgv[3 * U_DIM + u];

  v8f creg = {0.f, 0.f, 0.f, 0.f, 0.f, 0.f, 0.f, 0.f};

  int p = 0;
  for (int t = 0; t < T_LEN; ++t) {
    const bool hasNext = (t + 1 < T_LEN);

    // ---- issue t+1 input loads now; consumed after the WMMA chain ----
    float nx0 = 0.f, nx1 = 0.f;
    v8h  nv = {};
    if constexpr (FIRST) {
      if (hasNext) {
        const int row = tid >> 5, c = tid & 31;
        nx0 = xin[(((size_t)(wg * 16 + row    )) * T_LEN + (t + 1)) * F_IN + c];
        nx1 = xin[(((size_t)(wg * 16 + row + 8)) * T_LEN + (t + 1)) * F_IN + c];
      }
    } else {
      if (hasNext) {
        const _Float16* s1 = hseq + ((size_t)wg * T_LEN + (t + 1)) * 16 * U_DIM;
        nv = *(const v8h*)(s1 + tid * 8);
        if (tid < 32 && t + 2 < T_LEN)         // warm L2/WGP$ for t+2
          __builtin_prefetch((const void*)(s1 + 16 * U_DIM + tid * 64), 0, 1);
      }
    }

    // ---- z = [x_t | h_{t-1}] x [W;U] via WMMA (f32 accum) ----
    const _Float16* inRow = inBase + p * SLOT + colL * INST;
    const _Float16* hRow  = hBase + colL * HST;

    v8f acc[4];
#pragma unroll
    for (int gi = 0; gi < 4; ++gi)
      acc[gi] = (v8f){0.f, 0.f, 0.f, 0.f, 0.f, 0.f, 0.f, 0.f};

#pragma unroll
    for (int kc = 0; kc < KC; ++kc) {
      const _Float16* s = (kc < KCI) ? (inRow + kc * 32 + lb)
                                     : (hRow + (kc - KCI) * 32 + lb);
      v16h av = mk16(*(const v8h*)s, *(const v8h*)(s + 16));
#pragma unroll
      for (int gi = 0; gi < 4; ++gi) {         // nt = gate block + w
        const _Float16* fp = wlds + ((kc * 32 + (gi * 8 + w)) * 32 + lane) * 16;
        v16h bv = mk16(*(const v8h*)fp, *(const v8h*)(fp + 8));
        acc[gi] = __builtin_amdgcn_wmma_f32_16x16x32_f16(
            false, av, false, bv, (short)0, acc[gi], false, false);
      }
    }

    // Scheduling pipeline: per K-chunk there are 10 ds_load_b128 (2 A + 8 B)
    // and 4 WMMAs. Keep the loads one chunk ahead of the matrix ops so the
    // LDS latency hides behind WMMA execution (staggered s_wait_dscnt).
    __builtin_amdgcn_sched_group_barrier(SG_DSREAD, 10, 0);   // chunk 0 loads
#pragma unroll
    for (int kc = 0; kc < KC; ++kc) {
      if (kc + 1 < KC)
        __builtin_amdgcn_sched_group_barrier(SG_DSREAD, 10, 0); // chunk kc+1
      __builtin_amdgcn_sched_group_barrier(SG_MFMA, 4, 0);      // chunk kc
    }

    // ---- publish t+1 input into the alternate slot (load has drained) ----
    if (hasNext) {
      _Float16* dslot = inBase + (p ^ 1) * SLOT;
      if constexpr (FIRST) {
        const int row = tid >> 5, c = tid & 31;
        dslot[row * INST + c]       = (_Float16)nx0;
        dslot[(row + 8) * INST + c] = (_Float16)nx1;
      } else {
        *(v8h*)(dslot + (tid >> 4) * INST + (tid & 15) * 8) = nv;
      }
    }
    __syncthreads();   // all A reads done; t+1 input slot published

    // ---- gates, cell update, write new h (LDS + global h-seq) ----
    _Float16* hout = LAST ? (_Float16*)nullptr
                          : hseq + ((size_t)wg * T_LEN + t) * 16 * U_DIM + u;
#pragma unroll
    for (int r = 0; r < 8; ++r) {
      const int m = r + mOff;
      const float ig = sigmoid_f(acc[0][r] + bi);
      const float fg = sigmoid_f(acc[1][r] + bf);
      const float gg = tanh_f  (acc[2][r] + bg);
      const float og = sigmoid_f(acc[3][r] + bo);
      const float cn = fg * creg[r] + ig * gg;
      creg[r] = cn;
      const _Float16 hf = (_Float16)(og * tanh_f(cn));
      hBase[m * HST + u] = hf;                  // recurrent input for t+1
      if (!LAST) hout[m * U_DIM] = hf;          // next layer's input sequence
    }
    p ^= 1;
    __syncthreads();   // publish new h tile for next step's WMMA reads
  }
}

extern "C" __global__ void __launch_bounds__(256)
lstm3_fused_kernel(const float* __restrict__ x,
                   const float* __restrict__ W1, const float* __restrict__ U1,
                   const float* __restrict__ b1,
                   const float* __restrict__ W2, const float* __restrict__ U2,
                   const float* __restrict__ b2,
                   const float* __restrict__ W3, const float* __restrict__ U3,
                   const float* __restrict__ b3,
                   const float* __restrict__ Wd, const float* __restrict__ bd,
                   float* __restrict__ out, _Float16* __restrict__ hseq) {
  extern __shared__ _Float16 smem[];
  _Float16* wlds   = smem;                    // 8*32 frags * 1KB = 256 KB max
  _Float16* inBase = smem + 131072;           // 2 slots * 16*136 f16 = 8.5 KB
  _Float16* hBase  = smem + 131072 + 2 * SLOT;// 16*136 f16 = 4.25 KB

  lstm_layer<32,  true,  false>(x,       hseq, W1, U1, b1, wlds, inBase, hBase);
  __threadfence(); __syncthreads();           // h1 sequence visible within WG
  lstm_layer<128, false, false>(nullptr, hseq, W2, U2, b2, wlds, inBase, hBase);
  __threadfence(); __syncthreads();
  lstm_layer<128, false, true >(nullptr, hseq, W3, U3, b3, wlds, inBase, hBase);
  __syncthreads();

  // dense head: relu(h3 @ Wd + bd); h3 sits in the LDS h tile
  const int tid = threadIdx.x;
  if (tid < 16) {
    float acc = bd[0];
    const _Float16* hrow = hBase + tid * HST;
#pragma unroll 8
    for (int k = 0; k < 128; ++k) acc += (float)hrow[k] * Wd[k];
    out[blockIdx.x * 16 + tid] = fmaxf(acc, 0.0f);
  }
}

extern "C" void kernel_launch(void* const* d_in, const int* in_sizes, int n_in,
                              void* d_out, int out_size, void* d_ws, size_t ws_size,
                              hipStream_t stream) {
  (void)in_sizes; (void)n_in; (void)out_size; (void)ws_size;
  const float* x  = (const float*)d_in[0];
  const float* W1 = (const float*)d_in[1];
  const float* U1 = (const float*)d_in[2];
  const float* b1 = (const float*)d_in[3];
  const float* W2 = (const float*)d_in[4];
  const float* U2 = (const float*)d_in[5];
  const float* b2 = (const float*)d_in[6];
  const float* W3 = (const float*)d_in[7];
  const float* U3 = (const float*)d_in[8];
  const float* b3 = (const float*)d_in[9];
  const float* Wd = (const float*)d_in[10];
  const float* bd = (const float*)d_in[11];
  float*    out  = (float*)d_out;
  _Float16* hseq = (_Float16*)d_ws;           // needs 256*512*128*2 = 32 MiB

  const size_t smem_bytes =
      (size_t)(131072 + 2 * SLOT + 16 * HST) * sizeof(_Float16); // ~268.5 KB
  lstm3_fused_kernel<<<dim3(16), dim3(256), smem_bytes, stream>>>(
      x, W1, U1, b1, W2, U2, b2, W3, U3, b3, Wd, bd, out, hseq);
}
